// KernelCometric_55508157333788
// MI455X (gfx1250) — matrix operand order repaired and checked by reading, not compiled, not run on hardware
//
#include <hip/hip_runtime.h>
#include <hip/hip_bf16.h>
#include <hip/hip_fp16.h>

typedef _Float16 v16h __attribute__((ext_vector_type(16)));
typedef _Float16 h8   __attribute__((ext_vector_type(8)));
typedef float    v8f  __attribute__((ext_vector_type(8)));
typedef float    f4   __attribute__((ext_vector_type(4)));

#define REG_COEF 0.001f
#define D2OUT 4096   // d*d = 64*64

static __device__ __forceinline__ v16h cat16(h8 lo, h8 hi) {
  return __builtin_shufflevector(lo, hi, 0, 1, 2, 3, 4, 5, 6, 7,
                                         8, 9, 10, 11, 12, 13, 14, 15);
}

// ---------------------------------------------------------------------------
// Kernel 0: squared norms of q rows and c rows (d = 64)
// ---------------------------------------------------------------------------
__global__ void k0_sq_norms(const float* __restrict__ q, const float* __restrict__ c,
                            float* __restrict__ sqq, float* __restrict__ sqc,
                            int B, int K) {
  int i = blockIdx.x * blockDim.x + threadIdx.x;
  if (i < B) {
    const f4* p = (const f4*)(q + (size_t)i * 64);
    float s = 0.f;
#pragma unroll
    for (int j = 0; j < 16; ++j) { f4 v = p[j]; s += v.x*v.x + v.y*v.y + v.z*v.z + v.w*v.w; }
    sqq[i] = s;
  } else if (i < B + K) {
    int k = i - B;
    const f4* p = (const f4*)(c + (size_t)k * 64);
    float s = 0.f;
#pragma unroll
    for (int j = 0; j < 16; ++j) { f4 v = p[j]; s += v.x*v.x + v.y*v.y + v.z*v.z + v.w*v.w; }
    sqc[k] = s;
  }
}

// ---------------------------------------------------------------------------
// Kernel 0b: pack g_inv_c (K x 4096, f32 row-major over k) into f16 WMMA-B
// layout: Gp[(kblk*4096 + col)*16 + i] = g[(kblk*16+i)*4096 + col]
// One thread per (kblk, col) 32-byte chunk. Reads coalesced across col.
// ---------------------------------------------------------------------------
__global__ void k0b_pack_g(const float* __restrict__ g, _Float16* __restrict__ gp, int K) {
  int t    = blockIdx.x * blockDim.x + threadIdx.x;
  int col  = t & (D2OUT - 1);
  int kblk = t >> 12;
  if (kblk >= (K >> 4)) return;
  const float* src = g + (size_t)kblk * 16 * D2OUT + col;
  h8 lo, hi;
#pragma unroll
  for (int i = 0; i < 8; ++i) lo[i] = (_Float16)src[(size_t)i * D2OUT];
#pragma unroll
  for (int i = 0; i < 8; ++i) hi[i] = (_Float16)src[(size_t)(i + 8) * D2OUT];
  h8* dst = (h8*)(gp + ((size_t)kblk * D2OUT + col) * 16);
  dst[0] = lo;
  dst[1] = hi;
}

// ---------------------------------------------------------------------------
// Kernel 1: weights[b,k] = exp(-bw_k * max(|q_b|^2+|c_k|^2-2 q_b.c_k, 0))
// One wave per 16x16 tile of the (B x K) weight matrix; q.cT via WMMA f16,
// K-dim of the WMMA = d = 64 (2 steps of 32). Result stored as f16 row-major.
// ---------------------------------------------------------------------------
__global__ void k1_weights(const float* __restrict__ q, const float* __restrict__ c,
                           const float* __restrict__ bw,
                           const float* __restrict__ sqq, const float* __restrict__ sqc,
                           _Float16* __restrict__ W, int B, int K) {
  int wave = threadIdx.x >> 5;
  int lane = threadIdx.x & 31;
  int tile = blockIdx.x * 8 + wave;
  int ntK  = K >> 4;
  int ntot = (B >> 4) * ntK;
  if (tile >= ntot) return;                 // wave-uniform: EXEC stays all-ones
  int m0   = (tile / ntK) << 4;
  int n0   = (tile % ntK) << 4;
  int half = lane >> 4;                     // 0 or 1
  int mloc = lane & 15;

  v8f acc = {};
#pragma unroll
  for (int ks = 0; ks < 2; ++ks) {
    // A = q tile (16 x 32): lane holds K = 16*half + 0..15
    const float* arow = q + (size_t)(m0 + mloc) * 64 + ks * 32 + half * 16;
    v16h a;
#pragma unroll
    for (int i = 0; i < 8; ++i) a[i]     = (_Float16)arow[i];
#pragma unroll
    for (int i = 0; i < 8; ++i) a[i + 8] = (_Float16)arow[8 + i];
    // B = c^T tile (32 x 16): lane holds N = lane%16, K = 16*half + 0..15
    const float* bcol = c + (size_t)(n0 + mloc) * 64 + ks * 32 + half * 16;
    v16h b;
#pragma unroll
    for (int i = 0; i < 16; ++i) b[i] = (_Float16)bcol[i];
    acc = __builtin_amdgcn_wmma_f32_16x16x32_f16(false, a, false, b, (short)0, acc,
                                                 false, false);
  }

  int   ng   = n0 + mloc;
  float sqcn = sqc[ng];
  float bwn  = bw[ng];
#pragma unroll
  for (int r = 0; r < 8; ++r) {
    int   m  = m0 + r + half * 8;           // C layout: M = r + 8*(lane/16)
    float d2 = fmaxf(sqq[m] + sqcn - 2.0f * acc[r], 0.0f);
    float w  = __expf(-bwn * d2);
    W[(size_t)m * K + ng] = (_Float16)w;
  }
}

// ---------------------------------------------------------------------------
// Kernel 2: g_inv = W (B x K) @ Gpacked (K x 4096) + REG_COEF * I
// Register-blocked: one wave owns a 32x64 output block (2 M-tiles x 4 N-tiles
// = 8 f32 accumulators). Per k-step (32): 4 A loads + 8 B loads feed 8 WMMAs
// (0.75 KB operand traffic per WMMA instead of 2 KB unblocked).
// ---------------------------------------------------------------------------
__global__ void k2_ginv(const _Float16* __restrict__ W, const _Float16* __restrict__ Gp,
                        float* __restrict__ out, int B, int K) {
  int wave = threadIdx.x >> 5;
  int lane = threadIdx.x & 31;
  int job  = blockIdx.x * 8 + wave;
  const int NB = D2OUT >> 6;                // 64 n-blocks of 64 columns
  int njobs = (B >> 5) * NB;                // m-blocks of 32 rows
  if (job >= njobs) return;                 // wave-uniform
  int m0   = (job / NB) << 5;
  int n0   = (job % NB) << 6;
  int half = lane >> 4;
  int mloc = lane & 15;

  const _Float16* arow0 = W + (size_t)(m0 + mloc) * K + half * 16;
  const _Float16* arow1 = arow0 + (size_t)16 * K;

  v8f acc[2][4];
#pragma unroll
  for (int mt = 0; mt < 2; ++mt)
#pragma unroll
    for (int nt = 0; nt < 4; ++nt) acc[mt][nt] = (v8f){};

#pragma unroll 2
  for (int it = 0; it < K / 32; ++it) {
    // A fragments: W[m][it*32 + 16*half + {0..15}] -- 16B contiguous loads
    v16h a0 = cat16(*(const h8*)(arow0 + it * 32),
                    *(const h8*)(arow0 + it * 32 + 8));
    v16h a1 = cat16(*(const h8*)(arow1 + it * 32),
                    *(const h8*)(arow1 + it * 32 + 8));
    // B fragments: packed chunk for kblk = 2*it + half, cols n0 + nt*16 + mloc
    const _Float16* bbase = Gp + ((size_t)(it * 2 + half) * D2OUT + n0 + mloc) * 16;
#pragma unroll
    for (int nt = 0; nt < 4; ++nt) {
      const h8* bp = (const h8*)(bbase + (size_t)nt * 256);   // 16 cols * 16 halves
      v16h b = cat16(bp[0], bp[1]);
      acc[0][nt] = __builtin_amdgcn_wmma_f32_16x16x32_f16(false, a0, false, b,
                                                          (short)0, acc[0][nt],
                                                          false, false);
      acc[1][nt] = __builtin_amdgcn_wmma_f32_16x16x32_f16(false, a1, false, b,
                                                          (short)0, acc[1][nt],
                                                          false, false);
    }
  }

#pragma unroll
  for (int nt = 0; nt < 4; ++nt) {
    int   col  = n0 + nt * 16 + mloc;
    float diag = ((col >> 6) == (col & 63)) ? REG_COEF : 0.0f;
#pragma unroll
    for (int mt = 0; mt < 2; ++mt) {
#pragma unroll
      for (int r = 0; r < 8; ++r) {
        int m = m0 + mt * 16 + r + half * 8;   // C layout: M = r + 8*(lane/16)
        out[(size_t)m * D2OUT + col] = acc[mt][nt][r] + diag;
      }
    }
  }
}

// ---------------------------------------------------------------------------
extern "C" void kernel_launch(void* const* d_in, const int* in_sizes, int n_in,
                              void* d_out, int out_size, void* d_ws, size_t ws_size,
                              hipStream_t stream) {
  const float* q  = (const float*)d_in[0];
  const float* c  = (const float*)d_in[1];
  const float* bw = (const float*)d_in[2];
  const float* g  = (const float*)d_in[3];
  float* out = (float*)d_out;

  const int B = in_sizes[0] / 64;   // 8192
  const int K = in_sizes[2];        // 1024

  // workspace layout
  char*  ws   = (char*)d_ws;
  size_t off1 = (size_t)B * 4;                        // sqc after sqq
  size_t off2 = off1 + (size_t)K * 4;
  size_t offW = (off2 + 255) & ~(size_t)255;          // W f16: B*K
  size_t offG = offW + (size_t)B * K * 2;             // Gp f16: K*4096
  float*    sqq = (float*)ws;
  float*    sqc = (float*)(ws + off1);
  _Float16* W   = (_Float16*)(ws + offW);
  _Float16* Gp  = (_Float16*)(ws + offG);

  // 0: norms
  {
    int n = B + K;
    k0_sq_norms<<<(n + 255) / 256, 256, 0, stream>>>(q, c, sqq, sqc, B, K);
  }
  // 0b: pack g_inv_c into WMMA-B f16 layout
  {
    int n = (K >> 4) * D2OUT;
    k0b_pack_g<<<(n + 255) / 256, 256, 0, stream>>>(g, Gp, K);
  }
  // 1: weights (WMMA)
  {
    int tiles = (B >> 4) * (K >> 4);
    k1_weights<<<(tiles + 7) / 8, 256, 0, stream>>>(q, c, bw, sqq, sqc, W, B, K);
  }
  // 2: main GEMM (WMMA, 32x64 register-blocked per wave)
  {
    int jobs = (B >> 5) * (D2OUT >> 6);
    k2_ginv<<<(jobs + 7) / 8, 256, 0, stream>>>(W, Gp, out, B, K);
  }
}
